// ConvLSTM_84602265796699
// MI455X (gfx1250) — matrix operand order repaired and checked by reading, not compiled, not run on hardware
//
#include <hip/hip_runtime.h>
#include <hip/hip_bf16.h>

typedef __attribute__((ext_vector_type(16))) _Float16 v16h;
typedef __attribute__((ext_vector_type(8)))  _Float16 v8h;
typedef __attribute__((ext_vector_type(8)))  float    v8f;

__device__ __forceinline__ float sigmoidf_(float x) { return 1.0f / (1.0f + __expf(-x)); }

// ---------------- weight packing ----------------
// src: (K, N) f32 row-major. dst layout: for K-chunk kc (32 rows), column n,
// lane-half lh (0: K 0-15, 1: K 16-31): 16 contiguous halfs ->
// dst[((kc*N + n)*2 + lh)*16 + j] = src[(kc*32 + lh*16 + j)*N + n]  (zero if k>=K)
// A lane then fetches its whole B fragment with one 32-byte load.
template<int N>
__global__ void pack_weights(const float* __restrict__ s, _Float16* __restrict__ d, int K, int Kpad) {
  int i = blockIdx.x * blockDim.x + threadIdx.x;
  int total = Kpad * N;
  if (i >= total) return;
  int j  = i & 15;
  int lh = (i >> 4) & 1;
  int q  = i >> 5;            // kc*N + n
  int n  = q % N;
  int kc = q / N;
  int k  = kc * 32 + lh * 16 + j;
  d[i] = (k < K) ? (_Float16)s[(size_t)k * N + n] : (_Float16)0.0f;
}

__global__ void cvt_f32_to_f16(const float* __restrict__ s, _Float16* __restrict__ d, int n) {
  int i = blockIdx.x * blockDim.x + threadIdx.x;
  if (i < n) d[i] = (_Float16)s[i];
}

// ---------------- WMMA fragments ----------------
// A fragment (contiguous k order, same as packed B) from an LDS row of 32 halfs:
// lanes 0-15: K{0-15}; lanes 16-31: K{16-31}; two aligned b128 reads.
__device__ __forceinline__ v16h a_frag(const _Float16* atile, int row, int kb0) {
  const v8h* rp = (const v8h*)(atile + row * 32);
  v8h lo = rp[(kb0 >> 3) * 2];
  v8h hi = rp[(kb0 >> 3) * 2 + 1];
  v16h a;
#pragma unroll
  for (int j = 0; j < 8; ++j) { a[j] = lo[j]; a[8 + j] = hi[j]; }
  return a;
}

// ---------------- implicit-GEMM conv core ----------------
// 256 threads = 8 waves. Block tile: MTILE=64 pixels x 128 channels.
// Wave w owns N columns [16w,16w+16): 4 WMMAs per chunk sharing one B fragment.
// CIN==32 fast path: one K-chunk == one conv tap (ky,kx); A-stage is a single
// b128 global load + b128 LDS store per thread (channels are contiguous).
// LDS A tile is double-buffered: stage chunk kc+1 while WMMAs run on chunk kc,
// one barrier per chunk.
template<int H, int W, int CIN, int KHW, int PAD>
__device__ __forceinline__ void conv_accum_t(
    const _Float16* __restrict__ in, const _Float16* __restrict__ wpk,
    _Float16* atile /* 2 * 64*32 */, v8f* acc, int tid, int lane, int wave, int tile0, int M)
{
  constexpr int HW   = H * W;                    // power of two
  constexpr int K    = KHW * KHW * CIN;
  constexpr int KPAD = (K + 31) & ~31;
  constexpr int NCH  = KPAD / 32;
  int am  = lane & 15;
  int kb0 = (lane < 16) ? 0 : 8;
  int lh  = (lane < 16) ? 0 : 1;
  int n   = (wave << 4) + am;

  // staging coords: thread -> (row 0..63, 8 contiguous k)
  int srow = tid >> 2;
  int k0   = (tid & 3) << 3;
  int p    = tile0 + srow;
  int img  = p / HW;                              // shift (HW pow2)
  int rem  = p & (HW - 1);
  int oy   = rem / W;                             // shift
  int ox   = rem & (W - 1);

  auto stage = [&](int kc, _Float16* buf) {
    if constexpr (CIN == 32) {
      int ky = kc / KHW, kx = kc - (kc / KHW) * KHW;   // uniform, const divisor
      int iy = oy + ky - PAD, ix = ox + kx - PAD;
      uint4 v = {0u, 0u, 0u, 0u};
      if (p < M && iy >= 0 && iy < H && ix >= 0 && ix < W)
        v = *(const uint4*)(in + ((((size_t)img * H + iy) * W + ix) << 5) + k0);
      *(uint4*)(buf + srow * 32 + k0) = v;
    } else {
      // generic path (layer-1 input conv only, K=75, 3 chunks)
      for (int e = tid; e < 64 * 32; e += 256) {
        int m = e >> 5, k = e & 31;
        int kk = (kc << 5) + k;
        _Float16 v = (_Float16)0.0f;
        int pp = tile0 + m;
        if (pp < M && kk < K) {
          int im2 = pp / HW; int r2 = pp & (HW - 1);
          int oy2 = r2 / W,  ox2 = r2 & (W - 1);
          int ky = kk / (KHW * CIN); int rr = kk - ky * (KHW * CIN);
          int kx = rr / CIN;         int ci = rr - kx * CIN;
          int iy = oy2 + ky - PAD, ix = ox2 + kx - PAD;
          if (iy >= 0 && iy < H && ix >= 0 && ix < W)
            v = in[(((size_t)im2 * H + iy) * W + ix) * CIN + ci];
        }
        buf[e] = v;
      }
    }
  };

  stage(0, atile);
  __syncthreads();
  for (int kc = 0; kc < NCH; ++kc) {
    _Float16* cur = atile + ((kc & 1) << 11);
    if (kc + 1 < NCH) {
      stage(kc + 1, atile + (((kc + 1) & 1) << 11));
      __builtin_prefetch(wpk + ((((size_t)(kc + 1) * 128 + n) * 2 + lh) << 4), 0, 3);
    }
    v16h b = ((const v16h*)wpk)[((size_t)kc * 128 + n) * 2 + lh];
#pragma unroll
    for (int g = 0; g < 4; ++g) {
      v16h a = a_frag(cur, (g << 4) + am, kb0);
      acc[g] = __builtin_amdgcn_wmma_f32_16x16x32_f16(false, a, false, b, (short)0, acc[g], false, false);
    }
    __syncthreads();
  }
}

// input conv: out (M,128) f16 = conv(in, W) + bias
template<int H, int W, int CIN, int KHW, int PAD>
__global__ __launch_bounds__(256) void conv_in_t(
    const _Float16* __restrict__ in, const _Float16* __restrict__ wpk,
    const float* __restrict__ bias, _Float16* __restrict__ out, int Nimg)
{
  __shared__ __align__(16) _Float16 atile[2 * 64 * 32];
  int tid = threadIdx.x, lane = tid & 31, wave = tid >> 5;
  int M = Nimg * H * W;
  int tile0 = blockIdx.x << 6;
  v8f acc[4] = {};
  conv_accum_t<H, W, CIN, KHW, PAD>(in, wpk, atile, acc, tid, lane, wave, tile0, M);
  int n = (wave << 4) + (lane & 15);
  float bn = bias[n];
#pragma unroll
  for (int g = 0; g < 4; ++g)
#pragma unroll
    for (int r = 0; r < 8; ++r) {
      int m = (g << 4) + r + ((lane >> 4) << 3);
      int p = tile0 + m;
      if (p < M) out[((size_t)p << 7) + n] = (_Float16)(acc[g][r] + bn);
    }
}

// one ConvLSTM step: z = xz_t + conv(h,U); gates i,f,g,o; update c,h; emit hs_t
template<int H, int W, int KHW, int PAD>
__global__ __launch_bounds__(256) void step_t(
    const _Float16* __restrict__ h_in, const _Float16* __restrict__ upk,
    const _Float16* __restrict__ xz,   // (B*T, HW, 128)
    float* __restrict__ cbuf, _Float16* __restrict__ h_out,
    _Float16* __restrict__ hs,         // (B*T, HW, 32)
    int Bn, int t, int T)
{
  constexpr int HW = H * W;
  __shared__ __align__(16) _Float16 atile[2 * 64 * 32];
  __shared__ float zbuf[64 * 128];
  int tid = threadIdx.x, lane = tid & 31, wave = tid >> 5;
  int M = Bn * HW;
  int tile0 = blockIdx.x << 6;
  v8f acc[4] = {};
  conv_accum_t<H, W, 32, KHW, PAD>(h_in, upk, atile, acc, tid, lane, wave, tile0, M);
  int n = (wave << 4) + (lane & 15);
#pragma unroll
  for (int g = 0; g < 4; ++g)
#pragma unroll
    for (int r = 0; r < 8; ++r) {
      int m = (g << 4) + r + ((lane >> 4) << 3);
      int p = tile0 + m;
      float z = acc[g][r];
      if (p < M) {
        int b = p / HW; int rem = p & (HW - 1);
        z += (float)xz[(((size_t)(b * T + t) * HW + rem) << 7) + n];
      }
      zbuf[(m << 7) + n] = z;
    }
  __syncthreads();
  for (int idx = tid; idx < 64 * 32; idx += 256) {   // 64 pixels x 32 features
    int m = idx >> 5, f = idx & 31;
    int p = tile0 + m;
    if (p < M) {
      float zi = zbuf[(m << 7) +      f];
      float zf = zbuf[(m << 7) + 32 + f];
      float zg = zbuf[(m << 7) + 64 + f];
      float zo = zbuf[(m << 7) + 96 + f];
      float cn = sigmoidf_(zf) * cbuf[((size_t)p << 5) + f] + sigmoidf_(zi) * fmaxf(zg, 0.0f);
      float hv = sigmoidf_(zo) * fmaxf(cn, 0.0f);
      cbuf[((size_t)p << 5) + f]  = cn;
      h_out[((size_t)p << 5) + f] = (_Float16)hv;
      int b = p / HW; int rem = p & (HW - 1);
      hs[(((size_t)(b * T + t) * HW + rem) << 5) + f] = (_Float16)hv;
    }
  }
}

// BN (inference) + 2x2 maxpool, channels_last, F=32
template<int H, int W>
__global__ void bnpool_t(const _Float16* __restrict__ hs,
    const float* __restrict__ g, const float* __restrict__ be,
    const float* __restrict__ mean, const float* __restrict__ var,
    _Float16* __restrict__ out, int BT)
{
  constexpr int Ho = H >> 1, Wo = W >> 1;
  int Nout = BT * Ho * Wo * 32;
  int i = blockIdx.x * blockDim.x + threadIdx.x;
  if (i >= Nout) return;
  int f = i & 31; int q = i >> 5;
  int ox = q & (Wo - 1); q >>= 31 - __builtin_clz(Wo);
  int oy = q & (Ho - 1); int img = q >> (31 - __builtin_clz(Ho));
  float sc = g[f] * rsqrtf(var[f] + 1e-3f);
  float sh = be[f] - mean[f] * sc;
  size_t base = ((((size_t)img * H + 2 * oy) * W + 2 * ox) << 5) + f;
  float a0 = (float)hs[base] * sc + sh;
  float a1 = (float)hs[base + 32] * sc + sh;
  float a2 = (float)hs[base + (size_t)W * 32] * sc + sh;
  float a3 = (float)hs[base + (size_t)W * 32 + 32] * sc + sh;
  out[i] = (_Float16)fmaxf(fmaxf(a0, a1), fmaxf(a2, a3));
}

// dense(20480->64)+softmax+dense(64->10)+softmax; 4-wave block; WMMA GEMM.
// act16 has 16 rows (8 real, 8 zero-padded) -> unconditional b128 loads.
__global__ __launch_bounds__(128) void head_kernel(
    const _Float16* __restrict__ act16,  // (16,20480) f16, rows 8-15 zero
    const _Float16* __restrict__ w1p,    // packed (20480/32, 64, 2, 16) f16
    const float* __restrict__ b1,
    const float* __restrict__ w2,        // (64,10) f32
    const float* __restrict__ b2,
    float* __restrict__ out)             // (8,10) f32
{
  __shared__ float logits[8 * 64];
  __shared__ float probs[8 * 64];
  __shared__ float l2[80];
  int tid = threadIdx.x, lane = tid & 31, wave = tid >> 5;  // 4 waves -> 64 cols
  int n   = (wave << 4) + (lane & 15);
  int am  = lane & 15;
  int kb0 = (lane < 16) ? 0 : 8;
  int lh  = (lane < 16) ? 0 : 1;
  v8f acc = {};
  for (int kc = 0; kc < 640; ++kc) {   // K = 20480
    const v8h* ap = (const v8h*)(act16 + (size_t)am * 20480 + (kc << 5));
    v8h lo = ap[(kb0 >> 3) * 2];
    v8h hi = ap[(kb0 >> 3) * 2 + 1];
    v16h a;
#pragma unroll
    for (int j = 0; j < 8; ++j) { a[j] = lo[j]; a[8 + j] = hi[j]; }
    v16h b = ((const v16h*)w1p)[((size_t)kc * 64 + n) * 2 + lh];
    acc = __builtin_amdgcn_wmma_f32_16x16x32_f16(false, a, false, b, (short)0, acc, false, false);
  }
#pragma unroll
  for (int r = 0; r < 8; ++r) {
    int m = r + ((lane >> 4) << 3);
    if (m < 8) logits[m * 64 + n] = acc[r] + b1[n];
  }
  __syncthreads();
  if (tid < 8) {
    float mx = -1e30f;
    for (int k = 0; k < 64; ++k) mx = fmaxf(mx, logits[tid * 64 + k]);
    float s = 0.f;
    for (int k = 0; k < 64; ++k) { float e = __expf(logits[tid * 64 + k] - mx); probs[tid * 64 + k] = e; s += e; }
    float inv = 1.0f / s;
    for (int k = 0; k < 64; ++k) probs[tid * 64 + k] *= inv;
  }
  __syncthreads();
  if (tid < 80) {
    int m = tid / 10, j = tid - m * 10;
    float s = b2[j];
    for (int k = 0; k < 64; ++k) s += probs[m * 64 + k] * w2[k * 10 + j];
    l2[tid] = s;
  }
  __syncthreads();
  if (tid < 8) {
    float mx = -1e30f;
    for (int j = 0; j < 10; ++j) mx = fmaxf(mx, l2[tid * 10 + j]);
    float s = 0.f; float e[10];
    for (int j = 0; j < 10; ++j) { e[j] = __expf(l2[tid * 10 + j] - mx); s += e[j]; }
    float inv = 1.0f / s;
    for (int j = 0; j < 10; ++j) out[tid * 10 + j] = e[j] * inv;
  }
}

// ---------------- host orchestration ----------------
extern "C" void kernel_launch(void* const* d_in, const int* in_sizes, int n_in,
                              void* d_out, int out_size, void* d_ws, size_t ws_size,
                              hipStream_t stream) {
  (void)in_sizes; (void)n_in; (void)out_size; (void)ws_size;
  const float* x   = (const float*)d_in[0];
  const float* W1  = (const float*)d_in[1];
  const float* U1  = (const float*)d_in[2];
  const float* b1  = (const float*)d_in[3];
  const float* g1  = (const float*)d_in[4];
  const float* be1 = (const float*)d_in[5];
  const float* m1  = (const float*)d_in[6];
  const float* v1  = (const float*)d_in[7];
  const float* W2  = (const float*)d_in[8];
  const float* U2  = (const float*)d_in[9];
  const float* b2  = (const float*)d_in[10];
  const float* g2  = (const float*)d_in[11];
  const float* be2 = (const float*)d_in[12];
  const float* m2  = (const float*)d_in[13];
  const float* v2  = (const float*)d_in[14];
  const float* W3  = (const float*)d_in[15];
  const float* U3  = (const float*)d_in[16];
  const float* b3  = (const float*)d_in[17];
  const float* g3  = (const float*)d_in[18];
  const float* be3 = (const float*)d_in[19];
  const float* m3  = (const float*)d_in[20];
  const float* v3  = (const float*)d_in[21];
  const float* Wd1 = (const float*)d_in[22];
  const float* bd1 = (const float*)d_in[23];
  const float* Wd2 = (const float*)d_in[24];
  const float* bd2 = (const float*)d_in[25];

  char* ws = (char*)d_ws;
  size_t off = 0;
  auto alloc = [&](size_t bytes) -> void* {
    off = (off + 255) & ~(size_t)255;
    void* p = ws + off;
    off += bytes;
    return p;
  };

  _Float16* xh   = (_Float16*)alloc((size_t)983040 * 2);
  _Float16* w1q  = (_Float16*)alloc((size_t)96   * 128 * 2);   // Kpad 96
  _Float16* u1q  = (_Float16*)alloc((size_t)800  * 128 * 2);   // K=800 = 25*32
  _Float16* w2q  = (_Float16*)alloc((size_t)800  * 128 * 2);
  _Float16* u2q  = (_Float16*)alloc((size_t)800  * 128 * 2);
  _Float16* w3q  = (_Float16*)alloc((size_t)288  * 128 * 2);   // K=288 = 9*32
  _Float16* u3q  = (_Float16*)alloc((size_t)288  * 128 * 2);
  _Float16* wd1p = (_Float16*)alloc((size_t)20480 * 64 * 2);
  _Float16* xz1  = (_Float16*)alloc((size_t)80 * 4096 * 128 * 2);
  _Float16* hA1  = (_Float16*)alloc((size_t)8 * 4096 * 32 * 2);
  _Float16* hB1  = (_Float16*)alloc((size_t)8 * 4096 * 32 * 2);
  float*    c1b  = (float*)   alloc((size_t)8 * 4096 * 32 * 4);
  _Float16* hs1  = (_Float16*)alloc((size_t)80 * 4096 * 32 * 2);
  _Float16* p1   = (_Float16*)alloc((size_t)80 * 1024 * 32 * 2);
  _Float16* xz2  = (_Float16*)alloc((size_t)80 * 1024 * 128 * 2);
  _Float16* hA2  = (_Float16*)alloc((size_t)8 * 1024 * 32 * 2);
  _Float16* hB2  = (_Float16*)alloc((size_t)8 * 1024 * 32 * 2);
  float*    c2b  = (float*)   alloc((size_t)8 * 1024 * 32 * 4);
  _Float16* hs2  = (_Float16*)alloc((size_t)80 * 1024 * 32 * 2);
  _Float16* p2   = (_Float16*)alloc((size_t)80 * 256 * 32 * 2);
  _Float16* xz3  = (_Float16*)alloc((size_t)80 * 256 * 128 * 2);
  _Float16* hA3  = (_Float16*)alloc((size_t)8 * 256 * 32 * 2);
  _Float16* hB3  = (_Float16*)alloc((size_t)8 * 256 * 32 * 2);
  float*    c3b  = (float*)   alloc((size_t)8 * 256 * 32 * 4);
  _Float16* hs3  = (_Float16*)alloc((size_t)80 * 256 * 32 * 2);
  _Float16* act16= (_Float16*)alloc((size_t)16 * 20480 * 2);   // 8 real + 8 zero rows

  // conversions / packing
  cvt_f32_to_f16   <<<(983040 + 255) / 256, 256, 0, stream>>>(x, xh, 983040);
  pack_weights<128><<<(96   * 128 + 255) / 256, 256, 0, stream>>>(W1, w1q, 75, 96);
  pack_weights<128><<<(800  * 128 + 255) / 256, 256, 0, stream>>>(U1, u1q, 800, 800);
  pack_weights<128><<<(800  * 128 + 255) / 256, 256, 0, stream>>>(W2, w2q, 800, 800);
  pack_weights<128><<<(800  * 128 + 255) / 256, 256, 0, stream>>>(U2, u2q, 800, 800);
  pack_weights<128><<<(288  * 128 + 255) / 256, 256, 0, stream>>>(W3, w3q, 288, 288);
  pack_weights<128><<<(288  * 128 + 255) / 256, 256, 0, stream>>>(U3, u3q, 288, 288);
  pack_weights<64> <<<(20480 * 64 + 255) / 256, 256, 0, stream>>>(Wd1, wd1p, 20480, 20480);

  // ---- layer 1 (64x64, 5x5, Cin=3 -> F=32) ----
  conv_in_t<64, 64, 3, 5, 2><<<5120, 256, 0, stream>>>(xh, w1q, b1, xz1, 80);
  (void)hipMemsetAsync(hA1, 0, (size_t)8 * 4096 * 32 * 2, stream);
  (void)hipMemsetAsync(c1b, 0, (size_t)8 * 4096 * 32 * 4, stream);
  for (int t = 0; t < 10; ++t) {
    _Float16* hin  = (t & 1) ? hB1 : hA1;
    _Float16* hout = (t & 1) ? hA1 : hB1;
    step_t<64, 64, 5, 2><<<512, 256, 0, stream>>>(hin, u1q, xz1, c1b, hout, hs1, 8, t, 10);
  }
  bnpool_t<64, 64><<<(80 * 1024 * 32 + 255) / 256, 256, 0, stream>>>(hs1, g1, be1, m1, v1, p1, 80);

  // ---- layer 2 (32x32, 5x5, Cin=32 -> F=32) ----
  conv_in_t<32, 32, 32, 5, 2><<<1280, 256, 0, stream>>>(p1, w2q, b2, xz2, 80);
  (void)hipMemsetAsync(hA2, 0, (size_t)8 * 1024 * 32 * 2, stream);
  (void)hipMemsetAsync(c2b, 0, (size_t)8 * 1024 * 32 * 4, stream);
  for (int t = 0; t < 10; ++t) {
    _Float16* hin  = (t & 1) ? hB2 : hA2;
    _Float16* hout = (t & 1) ? hA2 : hB2;
    step_t<32, 32, 5, 2><<<128, 256, 0, stream>>>(hin, u2q, xz2, c2b, hout, hs2, 8, t, 10);
  }
  bnpool_t<32, 32><<<(80 * 256 * 32 + 255) / 256, 256, 0, stream>>>(hs2, g2, be2, m2, v2, p2, 80);

  // ---- layer 3 (16x16, 3x3, Cin=32 -> F=32) ----
  conv_in_t<16, 16, 32, 3, 1><<<320, 256, 0, stream>>>(p2, w3q, b3, xz3, 80);
  (void)hipMemsetAsync(hA3, 0, (size_t)8 * 256 * 32 * 2, stream);
  (void)hipMemsetAsync(c3b, 0, (size_t)8 * 256 * 32 * 4, stream);
  for (int t = 0; t < 10; ++t) {
    _Float16* hin  = (t & 1) ? hB3 : hA3;
    _Float16* hout = (t & 1) ? hA3 : hB3;
    step_t<16, 16, 3, 1><<<32, 256, 0, stream>>>(hin, u3q, xz3, c3b, hout, hs3, 8, t, 10);
  }
  // pooled layer-3 output lands in rows 0-7 of act16; rows 8-15 zeroed once
  (void)hipMemsetAsync(act16 + (size_t)8 * 20480, 0, (size_t)8 * 20480 * 2, stream);
  bnpool_t<16, 16><<<(80 * 64 * 32 + 255) / 256, 256, 0, stream>>>(hs3, g3, be3, m3, v3, act16, 80);

  // ---- head ----
  head_kernel<<<1, 128, 0, stream>>>(act16, wd1p, bd1, Wd2, bd2, (float*)d_out);
}
